// custom_gnn_3831110828326
// MI455X (gfx1250) — matrix operand order repaired and checked by reading, not compiled, not run on hardware
//
#include <hip/hip_runtime.h>
#include <math.h>

// ---------------- problem constants ----------------
#define NODES   4096
#define EDGES   131072
#define IN_DIM  384
#define HIDDEN  512
#define OUT_DIM 384
#define NLAYERS 3
#define NHEADS  4
#define FFDIM   2048
#define DHEAD   128

typedef __attribute__((ext_vector_type(16))) __bf16 bf16x16;
typedef __attribute__((ext_vector_type(8)))  float  f32x8;

#define WMMA_BF16(A_, B_, C_) \
  __builtin_amdgcn_wmma_f32_16x16x32_bf16(false, (A_), false, (B_), (short)0, (C_), false, false)

// ---------------- device helpers ----------------
__device__ __forceinline__ unsigned short f2bf(float f) {
  unsigned u = __float_as_uint(f);
  u += 0x7fffu + ((u >> 16) & 1u);      // round-to-nearest-even
  return (unsigned short)(u >> 16);
}
__device__ __forceinline__ unsigned pack2(float lo, float hi) {
  return (unsigned)f2bf(lo) | ((unsigned)f2bf(hi) << 16);
}
// order-preserving float <-> uint encoding for atomicMax on signed floats
__device__ __forceinline__ unsigned fenc(float f) {
  int i = __float_as_int(f);
  return (i >= 0) ? ((unsigned)i | 0x80000000u) : ~(unsigned)i;
}
__device__ __forceinline__ float fdec(unsigned u) {
  return (u & 0x80000000u) ? __int_as_float((int)(u & 0x7fffffffu))
                           : __int_as_float((int)~u);
}
__device__ __forceinline__ float gelu_exact(float x) {
  return 0.5f * x * (1.0f + erff(x * 0.70710678118654752440f));
}

// async global -> LDS copy, 16B per lane (CDNA5 ASYNCcnt path)
__device__ __forceinline__ void async_copy_b128(const void* gsrc, void* lds_dst) {
  const unsigned lds_off = (unsigned)(size_t)lds_dst;   // low 32 bits = LDS offset
  asm volatile("global_load_async_to_lds_b128 %0, %1, off"
               :: "v"(lds_off), "v"(gsrc)
               : "memory");
}
__device__ __forceinline__ void async_wait0() {
  asm volatile("s_wait_asynccnt 0" ::: "memory");
}

// ---------------- small utility kernels ----------------
__global__ __launch_bounds__(256) void cvt_bf16_k(const float* __restrict__ s,
                                                  unsigned short* __restrict__ d, int n) {
  int i = (blockIdx.x * 256 + threadIdx.x) * 4;
  if (i + 3 < n) {
    float4 f = *(const float4*)(s + i);
    uint2 o; o.x = pack2(f.x, f.y); o.y = pack2(f.z, f.w);
    *(uint2*)(d + i) = o;
  }
}

__global__ __launch_bounds__(256) void fill0_k(float* __restrict__ p, int n) {
  int i = blockIdx.x * 256 + threadIdx.x;
  if (i < n) p[i] = 0.0f;
}

// ---------------- fused residual + LayerNorm over HIDDEN=512 ----------------
__global__ __launch_bounds__(256) void add_ln_k(const float* __restrict__ x,
                                                const float* __restrict__ res,
                                                const float* __restrict__ g,
                                                const float* __restrict__ b,
                                                float* __restrict__ o) {
  const int row = blockIdx.x, t = threadIdx.x;
  const size_t base = (size_t)row * HIDDEN;
  float v0 = x[base + t];
  float v1 = x[base + 256 + t];
  if (res) { v0 += res[base + t]; v1 += res[base + 256 + t]; }
  float s = v0 + v1, q = v0 * v0 + v1 * v1;
#pragma unroll
  for (int m = 16; m >= 1; m >>= 1) {
    s += __shfl_xor(s, m, 32);
    q += __shfl_xor(q, m, 32);
  }
  __shared__ float sb[16];
  const int wv = t >> 5;
  if ((t & 31) == 0) { sb[wv] = s; sb[8 + wv] = q; }
  __syncthreads();
  if (t == 0) {
    float ts = 0.f, tq = 0.f;
#pragma unroll
    for (int i = 0; i < 8; ++i) { ts += sb[i]; tq += sb[8 + i]; }
    sb[0] = ts; sb[8] = tq;
  }
  __syncthreads();
  const float mean = sb[0] * (1.0f / HIDDEN);
  const float var  = sb[8] * (1.0f / HIDDEN) - mean * mean;
  const float inv  = rsqrtf(var + 1e-5f);
  o[base + t]        = (v0 - mean) * inv * g[t]        + b[t];
  o[base + 256 + t]  = (v1 - mean) * inv * g[256 + t]  + b[256 + t];
}

// ---------------- bf16 WMMA GEMM: C[M,Nc] = act(A[M,K] @ W[Nc,K]^T + bias) ----------------
// block = 256 threads (8 waves), tile 64x64, K-step 32.
// Weight tiles (already bf16) are staged with async global->LDS DMA (ASYNCcnt);
// activation tiles are converted fp32->bf16 on the fly through VGPRs.
template <int ACT>
__global__ __launch_bounds__(256) void gemm_wmma_k(const float* __restrict__ A,
                                                   const unsigned short* __restrict__ W,
                                                   const float* __restrict__ bias,
                                                   float* __restrict__ C,
                                                   int M, int Nc, int K) {
  __shared__ __align__(16) unsigned short As[64][40];   // 64x32 bf16, 80B rows (16B aligned)
  __shared__ __align__(16) unsigned short Bs[64][40];

  const int tid  = threadIdx.x;
  const int lane = tid & 31, wv = tid >> 5;
  const int half = lane >> 4, lm = lane & 15;
  const int tm = wv & 3, tn2 = wv >> 2;      // wave -> (M-tile, N-tile-pair)
  const int m0 = blockIdx.y * 64, n0 = blockIdx.x * 64;

  const int lrow = tid >> 2;                 // 0..63
  const int lcb  = (tid & 3) * 8;            // 0,8,16,24

  f32x8 acc0 = {0,0,0,0,0,0,0,0};
  f32x8 acc1 = {0,0,0,0,0,0,0,0};

  union Frag { bf16x16 v; unsigned u[8]; };

  const float*          arow_g = A + (size_t)(m0 + lrow) * K + lcb;
  const unsigned short* wrow_g = W + (size_t)(n0 + lrow) * K + lcb;

  for (int k0 = 0; k0 < K; k0 += 32) {
    __syncthreads();
    // W tile: async DMA straight into LDS (16B per lane)
    async_copy_b128(wrow_g + k0, &Bs[lrow][lcb]);
    // A tile: fp32 -> bf16 into LDS (coalesced 32B per thread)
    {
      const float4* ap = (const float4*)(arow_g + k0);
      float4 f0 = ap[0], f1 = ap[1];
      unsigned* drow = (unsigned*)&As[lrow][0];
      drow[(lcb >> 1) + 0] = pack2(f0.x, f0.y);
      drow[(lcb >> 1) + 1] = pack2(f0.z, f0.w);
      drow[(lcb >> 1) + 2] = pack2(f1.x, f1.y);
      drow[(lcb >> 1) + 3] = pack2(f1.z, f1.w);
    }
    if (k0 + 32 < K)  // hint next activation tile toward the caches
      __builtin_prefetch(arow_g + k0 + 32, 0, 1);
    async_wait0();            // our async B-tile writes have landed in LDS
    __syncthreads();          // publish across the workgroup

    Frag af, b0, b1;
    // A fragment: lane holds row lm, K contiguous 16 per half
    const unsigned* arow = (const unsigned*)&As[tm * 16 + lm][0];
#pragma unroll
    for (int j = 0; j < 4; ++j) af.u[j]     = arow[j + 8 * half];
#pragma unroll
    for (int j = 0; j < 4; ++j) af.u[4 + j] = arow[4 + j + 8 * half];
    // B fragments: lane holds col lm, K contiguous 16 per half
    const unsigned* brow0 = (const unsigned*)&Bs[tn2 * 32 + lm][0];
    const unsigned* brow1 = (const unsigned*)&Bs[tn2 * 32 + 16 + lm][0];
#pragma unroll
    for (int j = 0; j < 8; ++j) { b0.u[j] = brow0[8 * half + j]; b1.u[j] = brow1[8 * half + j]; }

    acc0 = WMMA_BF16(af.v, b0.v, acc0);
    acc1 = WMMA_BF16(af.v, b1.v, acc1);
  }

  // epilogue: C layout vgpr i <-> row i+8*half, col = lane&15
#pragma unroll
  for (int jt = 0; jt < 2; ++jt) {
    const int gn = n0 + (tn2 * 2 + jt) * 16 + lm;
    const float bb = bias ? bias[gn] : 0.0f;
    f32x8 a = jt ? acc1 : acc0;
#pragma unroll
    for (int i = 0; i < 8; ++i) {
      const int gm = m0 + tm * 16 + i + 8 * half;
      float c = a[i] + bb;
      if constexpr (ACT == 1) c = gelu_exact(c);
      C[(size_t)gm * Nc + gn] = c;
    }
  }
}

// ---------------- flash attention (global attn), HEADS=4, DH=128, N=4096 ----------------
// block = 256 threads (8 waves); each wave owns 16 query rows x full DH accumulator.
__global__ __launch_bounds__(256) void flash_attn_k(const float* __restrict__ qkv,
                                                    float* __restrict__ out) {
  __shared__ __align__(16) unsigned short Ks[32][136];   // keys row-major, padded
  __shared__ __align__(16) unsigned short Vt[128][40];   // V transposed [dim][key], padded
  __shared__ __align__(16) unsigned short Ps[8][16][40]; // per-wave P scratch (C->A bounce)

  const int tid  = threadIdx.x;
  const int lane = tid & 31, wv = tid >> 5;
  const int half = lane >> 4, lm = lane & 15;
  const int head = blockIdx.y;
  const int rowbase = blockIdx.x * 128 + wv * 16;

  union Frag { bf16x16 v; unsigned u[8]; };

  // persistent Q fragments (4 x K=32 chunks)
  Frag qf[4];
  {
    const float* qrow = qkv + (size_t)(rowbase + lm) * (3 * HIDDEN) + head * DHEAD;
#pragma unroll
    for (int kc = 0; kc < 4; ++kc) {
#pragma unroll
      for (int j = 0; j < 8; ++j) {
        const int k0 = kc * 32 + 2 * j + 16 * half;
        qf[kc].u[j] = pack2(qrow[k0], qrow[k0 + 1]);
      }
    }
  }

  f32x8 o[8];
#pragma unroll
  for (int j = 0; j < 8; ++j) o[j] = (f32x8){0,0,0,0,0,0,0,0};
  float mrow[8], lrow[8];
#pragma unroll
  for (int i = 0; i < 8; ++i) { mrow[i] = -1e30f; lrow[i] = 0.0f; }

  const float sc = 0.08838834764831845f;      // 1/sqrt(128)

  for (int nb = 0; nb < NODES / 32; ++nb) {
    __syncthreads();
    { // stage K and V^T tiles: 32 keys x 128 dims each
      const int r  = tid >> 3;                // key 0..31
      const int cb = (tid & 7) * 16;          // dim base
      const float* kp = qkv + (size_t)(nb * 32 + r) * (3 * HIDDEN) + HIDDEN + head * DHEAD + cb;
      const float* vp = kp + HIDDEN;
#pragma unroll
      for (int i = 0; i < 16; ++i) {
        Ks[r][cb + i]  = f2bf(kp[i]);
        Vt[cb + i][r]  = f2bf(vp[i]);
      }
    }
    __syncthreads();

    // S = Q @ K^T   (two 16x16 column tiles of keys)
    f32x8 s0 = {0,0,0,0,0,0,0,0};
    f32x8 s1 = {0,0,0,0,0,0,0,0};
    const unsigned* k0p = (const unsigned*)&Ks[lm][0];
    const unsigned* k1p = (const unsigned*)&Ks[16 + lm][0];
#pragma unroll
    for (int kc = 0; kc < 4; ++kc) {
      Frag b0, b1;
#pragma unroll
      for (int j = 0; j < 8; ++j) {
        b0.u[j] = k0p[kc * 16 + 8 * half + j];
        b1.u[j] = k1p[kc * 16 + 8 * half + j];
      }
      s0 = WMMA_BF16(qf[kc].v, b0.v, s0);
      s1 = WMMA_BF16(qf[kc].v, b1.v, s1);
    }

    // online softmax (row r = i + 8*half lives in vgpr i, replicated over 16 lanes)
    float corr[8];
#pragma unroll
    for (int i = 0; i < 8; ++i) { s0[i] *= sc; s1[i] *= sc; }
#pragma unroll
    for (int i = 0; i < 8; ++i) {
      float x = fmaxf(s0[i], s1[i]);
      x = fmaxf(x, __shfl_xor(x, 1, 32));
      x = fmaxf(x, __shfl_xor(x, 2, 32));
      x = fmaxf(x, __shfl_xor(x, 4, 32));
      x = fmaxf(x, __shfl_xor(x, 8, 32));
      const float mnew = fmaxf(mrow[i], x);
      const float p0 = expf(s0[i] - mnew);
      const float p1 = expf(s1[i] - mnew);
      Ps[wv][i + 8 * half][lm]      = f2bf(p0);
      Ps[wv][i + 8 * half][16 + lm] = f2bf(p1);
      float rs = p0 + p1;
      rs += __shfl_xor(rs, 1, 32);
      rs += __shfl_xor(rs, 2, 32);
      rs += __shfl_xor(rs, 4, 32);
      rs += __shfl_xor(rs, 8, 32);
      corr[i] = expf(mrow[i] - mnew);
      lrow[i] = lrow[i] * corr[i] + rs;
      mrow[i] = mnew;
    }
#pragma unroll
    for (int j = 0; j < 8; ++j)
#pragma unroll
      for (int i = 0; i < 8; ++i) o[j][i] *= corr[i];

    // bounce P through wave-private LDS to get A-fragment layout (DS ops are in-order per wave)
    Frag pf;
    const unsigned* prow = (const unsigned*)&Ps[wv][lm][0];
#pragma unroll
    for (int j = 0; j < 4; ++j) pf.u[j]     = prow[j + 8 * half];
#pragma unroll
    for (int j = 0; j < 4; ++j) pf.u[4 + j] = prow[4 + j + 8 * half];

    // O += P @ V
#pragma unroll
    for (int j = 0; j < 8; ++j) {
      Frag vb;
      const unsigned* vrow = (const unsigned*)&Vt[j * 16 + lm][0];
#pragma unroll
      for (int vv = 0; vv < 8; ++vv) vb.u[vv] = vrow[8 * half + vv];
      o[j] = WMMA_BF16(pf.v, vb.v, o[j]);
    }
  }

  // normalize + write
#pragma unroll
  for (int j = 0; j < 8; ++j) {
#pragma unroll
    for (int i = 0; i < 8; ++i) {
      const int gm = rowbase + i + 8 * half;
      const int gn = head * DHEAD + j * 16 + lm;
      out[(size_t)gm * HIDDEN + gn] = o[j][i] / lrow[i];
    }
  }
}

// ---------------- edge attention (segment softmax) ----------------
__global__ __launch_bounds__(256) void edge_logits_k(const float* __restrict__ q,
                                                     const float* __restrict__ k,
                                                     const float* __restrict__ ea,
                                                     const float* __restrict__ we,
                                                     const float* __restrict__ be,
                                                     const int* __restrict__ src,
                                                     const int* __restrict__ dst,
                                                     float* __restrict__ elog,
                                                     unsigned* __restrict__ mxu) {
  const int t = blockIdx.x * 256 + threadIdx.x;
  if (t >= EDGES * NHEADS) return;
  const int e = t >> 2, h = t & 3;
  const int s = src[e], d = dst[e];
  const float a = ea[e];
  const float* qp = q + (size_t)d * HIDDEN + h * DHEAD;
  const float* kp = k + (size_t)s * HIDDEN + h * DHEAD;
  const float* wp = we + h * DHEAD;
  const float* bp = be + h * DHEAD;
  float acc = 0.0f;
#pragma unroll 8
  for (int i = 0; i < DHEAD; ++i)
    acc += qp[i] * (kp[i] + a * wp[i] + bp[i]);
  acc *= 0.08838834764831845f;
  elog[t] = acc;
  atomicMax(&mxu[d * NHEADS + h], fenc(acc));
}

__global__ __launch_bounds__(256) void edge_exp_k(float* __restrict__ elog,
                                                  const unsigned* __restrict__ mxu,
                                                  float* __restrict__ den,
                                                  const int* __restrict__ dst) {
  const int t = blockIdx.x * 256 + threadIdx.x;
  if (t >= EDGES * NHEADS) return;
  const int e = t >> 2, h = t & 3;
  const int d = dst[e];
  const float a = expf(elog[t] - fdec(mxu[d * NHEADS + h]));
  elog[t] = a;
  atomicAdd(&den[d * NHEADS + h], a);
}

__global__ __launch_bounds__(256) void edge_msg_k(const float* __restrict__ v,
                                                  const float* __restrict__ ea,
                                                  const float* __restrict__ we,
                                                  const float* __restrict__ be,
                                                  const int* __restrict__ src,
                                                  const int* __restrict__ dst,
                                                  const float* __restrict__ elog,
                                                  const float* __restrict__ den,
                                                  float* __restrict__ agg) {
  const long long t = (long long)blockIdx.x * 256 + threadIdx.x;
  if (t >= (long long)EDGES * HIDDEN) return;
  const int e = (int)(t >> 9);
  const int c = (int)(t & 511);
  const int h = c >> 7;
  const int s = src[e], d = dst[e];
  const float alpha = elog[e * NHEADS + h] / fmaxf(den[d * NHEADS + h], 1e-16f);
  const float val = alpha * (v[(size_t)s * HIDDEN + c] + ea[e] * we[c] + be[c]);
  atomicAdd(&agg[(size_t)d * HIDDEN + c], val);
}

// ---------------- host orchestration ----------------
extern "C" void kernel_launch(void* const* d_in, const int* in_sizes, int n_in,
                              void* d_out, int out_size, void* d_ws, size_t ws_size,
                              hipStream_t stream) {
  (void)in_sizes; (void)n_in; (void)out_size; (void)ws_size;

  const float* embeddings = (const float*)d_in[0];
  const int*   eidx       = (const int*)d_in[1];
  const float* eattr      = (const float*)d_in[2];
  const float* in_w  = (const float*)d_in[3];
  const float* in_b  = (const float*)d_in[4];
  const float* wq    = (const float*)d_in[5];
  const float* bq    = (const float*)d_in[6];
  const float* wk    = (const float*)d_in[7];
  const float* bk    = (const float*)d_in[8];
  const float* wv_   = (const float*)d_in[9];
  const float* bv    = (const float*)d_in[10];
  const float* we    = (const float*)d_in[11];
  const float* be    = (const float*)d_in[12];
  const float* wo    = (const float*)d_in[13];
  const float* bo    = (const float*)d_in[14];
  const float* lln_s = (const float*)d_in[15];
  const float* lln_b = (const float*)d_in[16];
  const float* awqkv = (const float*)d_in[17];
  const float* abqkv = (const float*)d_in[18];
  const float* awo   = (const float*)d_in[19];
  const float* abo   = (const float*)d_in[20];
  const float* ln1_s = (const float*)d_in[21];
  const float* ln1_b = (const float*)d_in[22];
  const float* tff1w = (const float*)d_in[23];
  const float* tff1b = (const float*)d_in[24];
  const float* tff2w = (const float*)d_in[25];
  const float* tff2b = (const float*)d_in[26];
  const float* ln2_s = (const float*)d_in[27];
  const float* ln2_b = (const float*)d_in[28];
  const float* gln_s = (const float*)d_in[29];
  const float* gln_b = (const float*)d_in[30];
  const float* ff1w  = (const float*)d_in[31];
  const float* ff1b  = (const float*)d_in[32];
  const float* ff2w  = (const float*)d_in[33];
  const float* ff2b  = (const float*)d_in[34];
  const float* fln_s = (const float*)d_in[35];
  const float* fln_b = (const float*)d_in[36];
  const float* out_w = (const float*)d_in[37];
  const float* out_b = (const float*)d_in[38];

  const int* src = eidx;
  const int* dst = eidx + EDGES;

  // ---- workspace carve-out ----
  char* p = (char*)d_ws;
  auto alloc = [&](size_t bytes) -> char* {
    char* r = p;
    p += (bytes + 255) & ~(size_t)255;
    return r;
  };

  // bf16 weight pool
  const size_t SZ_INW  = (size_t)HIDDEN * IN_DIM;              // 196608
  const size_t SZ_SQ   = (size_t)HIDDEN * HIDDEN;              // 262144
  const size_t SZ_QKVW = (size_t)3 * HIDDEN * HIDDEN;          // 786432
  const size_t SZ_FFW  = (size_t)FFDIM * HIDDEN;               // 1048576
  const size_t SZ_OUTW = (size_t)OUT_DIM * HIDDEN;             // 196608
  const size_t WB_TOTAL = SZ_INW + NLAYERS * (4 * SZ_SQ + SZ_QKVW + SZ_SQ + 4 * SZ_FFW) + SZ_OUTW;

  unsigned short* WB = (unsigned short*)alloc(WB_TOTAL * sizeof(unsigned short));
  unsigned short* wb_inw  = WB;
  unsigned short* wb_wq   = wb_inw  + SZ_INW;
  unsigned short* wb_wk   = wb_wq   + NLAYERS * SZ_SQ;
  unsigned short* wb_wv   = wb_wk   + NLAYERS * SZ_SQ;
  unsigned short* wb_wo   = wb_wv   + NLAYERS * SZ_SQ;
  unsigned short* wb_aqkv = wb_wo   + NLAYERS * SZ_SQ;
  unsigned short* wb_awo  = wb_aqkv + NLAYERS * SZ_QKVW;
  unsigned short* wb_tff1 = wb_awo  + NLAYERS * SZ_SQ;
  unsigned short* wb_tff2 = wb_tff1 + NLAYERS * SZ_FFW;
  unsigned short* wb_ff1  = wb_tff2 + NLAYERS * SZ_FFW;
  unsigned short* wb_ff2  = wb_ff1  + NLAYERS * SZ_FFW;
  unsigned short* wb_outw = wb_ff2  + NLAYERS * SZ_FFW;

  float* hbuf = (float*)alloc((size_t)NODES * HIDDEN * 4);
  float* bufA = (float*)alloc((size_t)NODES * HIDDEN * 4);   // q / scratch
  float* bufB = (float*)alloc((size_t)NODES * HIDDEN * 4);   // k / scratch
  float* bufC = (float*)alloc((size_t)NODES * HIDDEN * 4);   // v / scratch
  float* bufD = (float*)alloc((size_t)NODES * HIDDEN * 4);   // agg / y
  float* qkvb = (float*)alloc((size_t)NODES * 3 * HIDDEN * 4);
  float* hidb = (float*)alloc((size_t)NODES * FFDIM * 4);
  unsigned* mxu = (unsigned*)alloc((size_t)NODES * NHEADS * 4);
  float* denb = (float*)alloc((size_t)NODES * NHEADS * 4);
  float* elog = (float*)alloc((size_t)EDGES * NHEADS * 4);

  // ---- launch helpers ----
  auto cvt = [&](const float* s, unsigned short* d, size_t n) {
    cvt_bf16_k<<<(unsigned)((n / 4 + 255) / 256), 256, 0, stream>>>(s, d, (int)n);
  };
  auto gemm = [&](const float* A, const unsigned short* W, const float* bias, float* C,
                  int M, int Nc, int K, int act) {
    dim3 g(Nc / 64, M / 64);
    if (act) gemm_wmma_k<1><<<g, 256, 0, stream>>>(A, W, bias, C, M, Nc, K);
    else     gemm_wmma_k<0><<<g, 256, 0, stream>>>(A, W, bias, C, M, Nc, K);
  };
  auto ln = [&](const float* x, const float* res, const float* g_, const float* b_, float* o_) {
    add_ln_k<<<NODES, 256, 0, stream>>>(x, res, g_, b_, o_);
  };
  auto zero = [&](float* ptr, size_t n) {
    fill0_k<<<(unsigned)((n + 255) / 256), 256, 0, stream>>>(ptr, (int)n);
  };

  // ---- convert all weights to bf16 once per launch ----
  cvt(in_w,  wb_inw,  SZ_INW);
  cvt(wq,    wb_wq,   NLAYERS * SZ_SQ);
  cvt(wk,    wb_wk,   NLAYERS * SZ_SQ);
  cvt(wv_,   wb_wv,   NLAYERS * SZ_SQ);
  cvt(wo,    wb_wo,   NLAYERS * SZ_SQ);
  cvt(awqkv, wb_aqkv, NLAYERS * SZ_QKVW);
  cvt(awo,   wb_awo,  NLAYERS * SZ_SQ);
  cvt(tff1w, wb_tff1, NLAYERS * SZ_FFW);
  cvt(tff2w, wb_tff2, NLAYERS * SZ_FFW);
  cvt(ff1w,  wb_ff1,  NLAYERS * SZ_FFW);
  cvt(ff2w,  wb_ff2,  NLAYERS * SZ_FFW);
  cvt(out_w, wb_outw, SZ_OUTW);

  // ---- input projection ----
  gemm(embeddings, wb_inw, in_b, hbuf, NODES, HIDDEN, IN_DIM, 0);

  for (int l = 0; l < NLAYERS; ++l) {
    const unsigned short* Wq   = wb_wq   + (size_t)l * SZ_SQ;
    const unsigned short* Wk   = wb_wk   + (size_t)l * SZ_SQ;
    const unsigned short* Wv   = wb_wv   + (size_t)l * SZ_SQ;
    const unsigned short* Wo   = wb_wo   + (size_t)l * SZ_SQ;
    const unsigned short* Wqkv = wb_aqkv + (size_t)l * SZ_QKVW;
    const unsigned short* Wawo = wb_awo  + (size_t)l * SZ_SQ;
    const unsigned short* Wt1  = wb_tff1 + (size_t)l * SZ_FFW;
    const unsigned short* Wt2  = wb_tff2 + (size_t)l * SZ_FFW;
    const unsigned short* Wf1  = wb_ff1  + (size_t)l * SZ_FFW;
    const unsigned short* Wf2  = wb_ff2  + (size_t)l * SZ_FFW;
    const float* we_l = we + (size_t)l * HIDDEN;
    const float* be_l = be + (size_t)l * HIDDEN;

    // --- GTConv ---
    gemm(hbuf, Wq, bq + l * HIDDEN, bufA, NODES, HIDDEN, HIDDEN, 0);  // q
    gemm(hbuf, Wk, bk + l * HIDDEN, bufB, NODES, HIDDEN, HIDDEN, 0);  // k
    gemm(hbuf, Wv, bv + l * HIDDEN, bufC, NODES, HIDDEN, HIDDEN, 0);  // v
    zero((float*)mxu, (size_t)NODES * NHEADS);                        // encoded -inf
    zero(denb, (size_t)NODES * NHEADS);
    zero(bufD, (size_t)NODES * HIDDEN);                               // agg
    edge_logits_k<<<(EDGES * NHEADS + 255) / 256, 256, 0, stream>>>(
        bufA, bufB, eattr, we_l, be_l, src, dst, elog, mxu);
    edge_exp_k<<<(EDGES * NHEADS + 255) / 256, 256, 0, stream>>>(elog, mxu, denb, dst);
    edge_msg_k<<<(unsigned)(((long long)EDGES * HIDDEN + 255) / 256), 256, 0, stream>>>(
        bufC, eattr, we_l, be_l, src, dst, elog, denb, bufD);
    gemm(bufD, Wo, bo + l * HIDDEN, bufA, NODES, HIDDEN, HIDDEN, 0);  // h_local
    ln(hbuf, bufA, lln_s + l * HIDDEN, lln_b + l * HIDDEN, hbuf);     // h = LN(h + h_local)

    // --- global transformer encoder layer ---
    gemm(hbuf, Wqkv, abqkv + l * 3 * HIDDEN, qkvb, NODES, 3 * HIDDEN, HIDDEN, 0);
    flash_attn_k<<<dim3(NODES / 128, NHEADS), 256, 0, stream>>>(qkvb, bufB);
    gemm(bufB, Wawo, abo + l * HIDDEN, bufC, NODES, HIDDEN, HIDDEN, 0);
    ln(hbuf, bufC, ln1_s + l * HIDDEN, ln1_b + l * HIDDEN, bufD);     // y
    gemm(bufD, Wt1, tff1b + l * FFDIM, hidb, NODES, FFDIM, HIDDEN, 1);   // GELU
    gemm(hidb, Wt2, tff2b + l * HIDDEN, bufA, NODES, HIDDEN, FFDIM, 0);
    ln(bufD, bufA, ln2_s + l * HIDDEN, ln2_b + l * HIDDEN, bufB);     // h_global
    ln(hbuf, bufB, gln_s + l * HIDDEN, gln_b + l * HIDDEN, hbuf);     // h = LN(h + h_global)

    // --- block FFN ---
    gemm(hbuf, Wf1, ff1b + l * FFDIM, hidb, NODES, FFDIM, HIDDEN, 1);    // GELU
    gemm(hidb, Wf2, ff2b + l * HIDDEN, bufA, NODES, HIDDEN, FFDIM, 0);
    ln(hbuf, bufA, fln_s + l * HIDDEN, fln_b + l * HIDDEN, hbuf);     // h = LN(h + ff)
  }

  // ---- output projection ----
  gemm(hbuf, wb_outw, out_b, (float*)d_out, NODES, OUT_DIM, HIDDEN, 0);
}
